// RoformerMHSAV1_56727928045876
// MI455X (gfx1250) — compile-verified, matrix operand order checked by reading, not run on hardware
//
#include <hip/hip_runtime.h>
#include <hip/hip_bf16.h>

// ---------------------------------------------------------------------------
// Roformer MHSA forward for MI455X (gfx1250, wave32, WMMA + TDM).
// B=8, T=1024, D=1024, H=16, DH=64.
// bf16 WMMA (16x16x32) everywhere, f32 accumulation.
// GEMMs: per-wave 32x64 register tile (8 accumulators), B panel staged into
// LDS by the Tensor Data Mover (tensor_load_to_lds + s_wait_tensorcnt).
// ---------------------------------------------------------------------------

typedef __bf16 bf16_t;
typedef __attribute__((ext_vector_type(16))) __bf16 v16bf;
typedef __attribute__((ext_vector_type(8)))  __bf16 v8bf;
typedef __attribute__((ext_vector_type(8)))  float  v8f;
typedef __attribute__((ext_vector_type(4)))  unsigned int u32x4;
typedef __attribute__((ext_vector_type(8)))  int i32x8;
typedef __attribute__((ext_vector_type(4)))  int i32x4;

#define CB   8
#define CT   1024
#define CD   1024
#define CH   16
#define CDH  64

#ifndef ATHENA_TDM_ARITY
#if defined(__clang_major__) && __clang_major__ >= 23
#define ATHENA_TDM_ARITY 6
#else
#define ATHENA_TDM_ARITY 5
#endif
#endif

__device__ __forceinline__ bf16_t f2bf(float f) {
  unsigned u = __builtin_bit_cast(unsigned, f);
  unsigned r = u + 0x7FFFu + ((u >> 16) & 1u);   // RNE to bf16
  unsigned short h = (unsigned short)(r >> 16);
  return __builtin_bit_cast(bf16_t, h);
}

// 16x32 bf16 WMMA A/B fragment from K-major storage (ISA 7.12.2):
//   lanes 0-15 : M = lane,    K = 0..7 / 16..23
//   lanes 16-31: M = lane-16, K = 8..15 / 24..31
// caller passes p0 = base + m*ldk + kb  (kb = 0 or 8 per lane half).
__device__ __forceinline__ v16bf load_frag_b16(const bf16_t* p0) {
  v8bf lo = *(const v8bf*)(p0);
  v8bf hi = *(const v8bf*)(p0 + 16);
  return __builtin_shufflevector(lo, hi, 0,1,2,3,4,5,6,7,8,9,10,11,12,13,14,15);
}

__device__ __forceinline__ v8f wmma_bf16(v16bf a, v16bf b, v8f c) {
  return __builtin_amdgcn_wmma_f32_16x16x32_bf16(
      false, a, false, b, (short)0, c, false, false);
}

// ---------------------------------------------------------------------------
// Stage a 64-row x 64-element bf16 tile (row stride ldk elements) into LDS.
// Preferred path: Tensor Data Mover. D# built per CDNA5 ISA ch.8:
//  group0: count=1 | lds_addr | global_addr[56:0] | type=2
//  group1: data_size=1(2B), tensor_dim0=ldk, tensor_dim1=rows,
//          tile_dim0=64 (contiguous K), tile_dim1=64, tensor_dim0_stride=ldk
// Issued by wave 0 only; caller must __syncthreads() after.
// ---------------------------------------------------------------------------
__device__ __forceinline__ void stage_b_tile(bf16_t* lds, const bf16_t* gsrc,
                                             int ldk, int rows,
                                             int tid, int wave) {
#if __has_builtin(__builtin_amdgcn_tensor_load_to_lds)
  if (wave == 0) {
    unsigned long long ga = (unsigned long long)(uintptr_t)gsrc;
    unsigned la = (unsigned)(uintptr_t)lds;     // LDS byte offset = addr[31:0]
    u32x4 g0 = {};
    g0.x = 1u;                                   // count=1 (user descriptor)
    g0.y = la;                                   // lds_addr
    g0.z = (unsigned)(ga & 0xFFFFFFFFull);       // global_addr[31:0]
    g0.w = (unsigned)((ga >> 32) & 0x1FFFFFFull) // global_addr[56:32]
         | (2u << 30);                           // type=2 ("image")
    i32x8 g1 = {};
    g1[0] = (int)(1u << 16);                     // data_size=1 -> 2 bytes
    g1[1] = (int)(((unsigned)ldk & 0xFFFFu) << 16);      // tensor_dim0[15:0]
    g1[2] = (int)((((unsigned)ldk >> 16) & 0xFFFFu)      // tensor_dim0[31:16]
                  | (((unsigned)rows & 0xFFFFu) << 16)); // tensor_dim1[15:0]
    g1[3] = (int)((((unsigned)rows >> 16) & 0xFFFFu)     // tensor_dim1[31:16]
                  | (64u << 16));                        // tile_dim0 = 64
    g1[4] = (int)64u;                            // tile_dim1 = 64 (tile_dim2=0)
    g1[5] = (int)(unsigned)ldk;                  // tensor_dim0_stride[31:0]
    g1[6] = 0;                                   // stride hi / dim1_stride lo
    g1[7] = 0;
    i32x4 g2 = {};
    i32x4 g3 = {};
#if ATHENA_TDM_ARITY == 6
    i32x8 g4 = {};
    __builtin_amdgcn_tensor_load_to_lds(g0, g1, g2, g3, g4, 0);
#else
    __builtin_amdgcn_tensor_load_to_lds(g0, g1, g2, g3, 0);
#endif
    __builtin_amdgcn_s_wait_tensorcnt(0);
  }
#else
  // Cooperative fallback: 512 x v8bf chunks over 128 threads.
  (void)wave; (void)rows;
  for (int i = tid; i < 64 * 8; i += 128) {
    int r = i >> 3, c = (i & 7) * 8;
    *(v8bf*)&lds[r * 64 + c] = *(const v8bf*)&gsrc[(size_t)r * ldk + c];
  }
#endif
}

// ---------------------------------------------------------------------------
// Kernel 0: transpose + fp32->bf16 cast (K x N row-major -> N x K K-major).
// ---------------------------------------------------------------------------
__global__ void __launch_bounds__(256)
k_transpose_cast(const float* __restrict__ W, bf16_t* __restrict__ Wt,
                 int Kdim, int N) {
  size_t idx = (size_t)blockIdx.x * blockDim.x + threadIdx.x;
  if (idx < (size_t)Kdim * N) {
    int n = (int)(idx / Kdim);
    int k = (int)(idx % Kdim);
    Wt[idx] = f2bf(W[(size_t)k * N + n]);
  }
}

// ---------------------------------------------------------------------------
// Kernel 1: LayerNorm over D=1024 -> bf16 Xn.
// ---------------------------------------------------------------------------
__global__ void __launch_bounds__(256)
k_layernorm(const float* __restrict__ x, const float* __restrict__ gamma,
            const float* __restrict__ beta, bf16_t* __restrict__ Xn) {
  __shared__ float red[256];
  __shared__ float mu_s, rstd_s;
  const int row = blockIdx.x;
  const int tid = threadIdx.x;
  const float* xr = x + (size_t)row * CD;

  float v[4];
  float s = 0.0f;
  #pragma unroll
  for (int i = 0; i < 4; ++i) { v[i] = xr[tid + 256 * i]; s += v[i]; }
  red[tid] = s; __syncthreads();
  for (int st = 128; st > 0; st >>= 1) {
    if (tid < st) red[tid] += red[tid + st];
    __syncthreads();
  }
  if (tid == 0) mu_s = red[0] * (1.0f / CD);
  __syncthreads();
  const float mu = mu_s;

  float s2 = 0.0f;
  #pragma unroll
  for (int i = 0; i < 4; ++i) { float d = v[i] - mu; s2 += d * d; }
  red[tid] = s2; __syncthreads();
  for (int st = 128; st > 0; st >>= 1) {
    if (tid < st) red[tid] += red[tid + st];
    __syncthreads();
  }
  if (tid == 0) rstd_s = rsqrtf(red[0] * (1.0f / CD) + 1e-5f);
  __syncthreads();
  const float rstd = rstd_s;

  #pragma unroll
  for (int i = 0; i < 4; ++i) {
    int c = tid + 256 * i;
    Xn[(size_t)row * CD + c] = f2bf((v[i] - mu) * rstd * gamma[c] + beta[c]);
  }
}

// ---------------------------------------------------------------------------
// Kernel 2: QKV GEMM (8192x3072x1024) + bias + RoPE.
// Block = 128 thr = 4 waves; block tile 128(M) x 64(N); wave tile 32x64.
// B panel (64 N-rows x 64 K) staged via TDM into LDS, shared by all waves.
// ---------------------------------------------------------------------------
__global__ void __launch_bounds__(128)
k_qkv_gemm_rope(const bf16_t* __restrict__ Xn, const bf16_t* __restrict__ WqkvT,
                const float* __restrict__ bqkv,
                bf16_t* __restrict__ Q, bf16_t* __restrict__ Kb,
                bf16_t* __restrict__ Vt) {
  __shared__ __align__(16) bf16_t ldsB[64 * 64];   // 8 KB
  const int tid  = threadIdx.x;
  const int lane = tid & 31;
  const int wave = tid >> 5;
  const int m0 = blockIdx.y * 128 + wave * 32;
  const int n0 = blockIdx.x * 64;
  const int mrow = lane & 15;
  const int hf   = lane >> 4;
  const int kb   = hf * 8;
  const int nloc = lane & 15;

  const bf16_t* aBase0 = Xn + (size_t)(m0 + mrow) * CD + kb;
  const bf16_t* aBase1 = aBase0 + (size_t)16 * CD;
  const bf16_t* bPanel = WqkvT + (size_t)n0 * CD;

  v8f acc[8];
  #pragma unroll
  for (int i = 0; i < 8; ++i) { v8f z = {}; acc[i] = z; }

  for (int kc = 0; kc < CD; kc += 64) {
    stage_b_tile(ldsB, bPanel + kc, CD, 3 * CD - n0, tid, wave);
    __syncthreads();
    __builtin_prefetch(aBase0 + kc + 64, 0, 0);   // global_prefetch_b8
    #pragma unroll
    for (int kl = 0; kl < 64; kl += 32) {
      v16bf a0 = load_frag_b16(aBase0 + kc + kl);
      v16bf a1 = load_frag_b16(aBase1 + kc + kl);
      #pragma unroll
      for (int nt = 0; nt < 4; ++nt) {
        v16bf bv = load_frag_b16(&ldsB[(nt * 16 + mrow) * 64 + kl + kb]);
        acc[nt * 2 + 0] = wmma_bf16(a0, bv, acc[nt * 2 + 0]);
        acc[nt * 2 + 1] = wmma_bf16(a1, bv, acc[nt * 2 + 1]);
      }
    }
    __syncthreads();
  }

  // Epilogue: C layout -> N = lane&15, M = r + 8*hf. RoPE over full D.
  #pragma unroll
  for (int nt = 0; nt < 4; ++nt) {
    const int col = n0 + nt * 16 + nloc;   // 0..3071
    const int sec = col >> 10;             // 0=q, 1=k, 2=v
    const int cs  = col & (CD - 1);
    const int i2  = cs >> 1;
    const float inv_freq =
        __powf(100000.0f, (-2.0f / (float)CD) * (float)(i2 - 1));
    const bool  evn  = (cs & 1) == 0;
    const float bias = bqkv[col];
    const int h  = cs >> 6;
    const int dh = cs & (CDH - 1);
    #pragma unroll
    for (int mt = 0; mt < 2; ++mt) {
      #pragma unroll
      for (int r = 0; r < 8; ++r) {
        const int row  = m0 + mt * 16 + r + 8 * hf;   // b*T + t
        const int t    = row & (CT - 1);
        const int bidx = row >> 10;
        float val = acc[nt * 2 + mt][r] + bias;
        float pv  = __shfl_xor(val, 1, 32);           // partner col^1 = lane^1
        float sn, csn;
        __sincosf((float)t * inv_freq, &sn, &csn);
        float out = evn ? (val * csn - pv * sn) : (val * csn + pv * sn);
        bf16_t ob = f2bf(out);
        const size_t bh = (size_t)bidx * CH + h;
        if (sec == 0)      Q [(bh * CT + t) * CDH + dh] = ob;
        else if (sec == 1) Kb[(bh * CT + t) * CDH + dh] = ob;
        else               Vt[(bh * CDH + dh) * CT + t] = ob;
      }
    }
  }
}

// ---------------------------------------------------------------------------
// Kernel 3: flash attention. 4 independent waves/block, each owns a 16-row
// query tile; keys streamed 32 at a time with online softmax.
// P transposed C->A layout through a private LDS slab (wave-internal, LDS
// ops are in-order per wave -> wave_barrier only, no workgroup barrier).
// ---------------------------------------------------------------------------
__global__ void __launch_bounds__(128)
k_attention(const bf16_t* __restrict__ Q, const bf16_t* __restrict__ K,
            const bf16_t* __restrict__ Vt, const unsigned char* __restrict__ smask,
            bf16_t* __restrict__ O) {
  __shared__ __align__(16) bf16_t Pl[4][16][32];

  const int lane = threadIdx.x & 31;
  const int wave = threadIdx.x >> 5;
  const int q0 = blockIdx.x * 64 + wave * 16;
  const int bh = blockIdx.y;                    // b*H + h
  const int b = bh >> 4, h = bh & 15;
  const int mrow = lane & 15;
  const int hf   = lane >> 4;
  const int kb   = hf * 8;
  const int nloc = lane & 15;

  const bf16_t* Qb = Q  + (size_t)bh * CT * CDH;
  const bf16_t* Kp = K  + (size_t)bh * CT * CDH;
  const bf16_t* Vb = Vt + (size_t)bh * CDH * CT;
  const unsigned char* mk = smask + (size_t)b * CT;
  bf16_t* Pw = &Pl[wave][0][0];

  const v16bf aq0 = load_frag_b16(Qb + (size_t)(q0 + mrow) * CDH + kb);
  const v16bf aq1 = load_frag_b16(Qb + (size_t)(q0 + mrow) * CDH + kb + 32);

  float mrun[8], lrun[8];
  v8f o[4];
  #pragma unroll
  for (int r = 0; r < 8; ++r) { mrun[r] = -1e30f; lrun[r] = 0.0f; }
  #pragma unroll
  for (int i = 0; i < 4; ++i) { v8f z = {}; o[i] = z; }

  for (int s0 = 0; s0 < CT; s0 += 32) {
    v8f sc[2];
    #pragma unroll
    for (int j = 0; j < 2; ++j) {
      const bf16_t* kbase = Kp + (size_t)(s0 + 16 * j + mrow) * CDH + kb;
      v16bf b0 = load_frag_b16(kbase);
      v16bf b1 = load_frag_b16(kbase + 32);
      v8f c = {};
      c = wmma_bf16(aq0, b0, c);
      c = wmma_bf16(aq1, b1, c);
      const int key = s0 + 16 * j + nloc;
      const float madd = mk[key] ? 0.0f : -1e30f;
      #pragma unroll
      for (int r = 0; r < 8; ++r) sc[j][r] = c[r] * 0.125f + madd;
    }

    #pragma unroll
    for (int r = 0; r < 8; ++r) {
      float mt = fmaxf(sc[0][r], sc[1][r]);
      #pragma unroll
      for (int d = 1; d < 16; d <<= 1) mt = fmaxf(mt, __shfl_xor(mt, d, 32));
      const float mnew  = fmaxf(mrun[r], mt);
      const float alpha = __expf(mrun[r] - mnew);
      const float p0 = __expf(sc[0][r] - mnew);
      const float p1 = __expf(sc[1][r] - mnew);
      float ps = p0 + p1;
      #pragma unroll
      for (int d = 1; d < 16; d <<= 1) ps += __shfl_xor(ps, d, 32);
      lrun[r] = lrun[r] * alpha + ps;
      mrun[r] = mnew;
      o[0][r] *= alpha; o[1][r] *= alpha; o[2][r] *= alpha; o[3][r] *= alpha;
      Pw[(r + 8 * hf) * 32 + nloc]      = f2bf(p0);
      Pw[(r + 8 * hf) * 32 + 16 + nloc] = f2bf(p1);
    }
    __builtin_amdgcn_wave_barrier();   // keep ds_store -> ds_load ordered

    v16bf ap = load_frag_b16(&Pw[mrow * 32 + kb]);
    #pragma unroll
    for (int nt = 0; nt < 4; ++nt) {
      v16bf bv = load_frag_b16(Vb + (size_t)(nt * 16 + mrow) * CT + s0 + kb);
      o[nt] = wmma_bf16(ap, bv, o[nt]);
    }
    __builtin_amdgcn_wave_barrier();
  }

  #pragma unroll
  for (int r = 0; r < 8; ++r) {
    const float inv = 1.0f / lrun[r];
    const int t = q0 + r + 8 * hf;
    #pragma unroll
    for (int nt = 0; nt < 4; ++nt) {
      const int colD = h * CDH + nt * 16 + nloc;
      O[((size_t)b * CT + t) * CD + colD] = f2bf(o[nt][r] * inv);
    }
  }
}

// ---------------------------------------------------------------------------
// Kernel 4: output projection (8192x1024x1024) + bias -> fp32.
// Same blocking as kernel 2 (TDM-staged B panel, 32x64 wave tiles).
// ---------------------------------------------------------------------------
__global__ void __launch_bounds__(128)
k_out_gemm(const bf16_t* __restrict__ A, const bf16_t* __restrict__ WoutT,
           const float* __restrict__ bout, float* __restrict__ out) {
  __shared__ __align__(16) bf16_t ldsB[64 * 64];
  const int tid  = threadIdx.x;
  const int lane = tid & 31;
  const int wave = tid >> 5;
  const int m0 = blockIdx.y * 128 + wave * 32;
  const int n0 = blockIdx.x * 64;
  const int mrow = lane & 15;
  const int hf   = lane >> 4;
  const int kb   = hf * 8;

  const bf16_t* aBase0 = A + (size_t)(m0 + mrow) * CD + kb;
  const bf16_t* aBase1 = aBase0 + (size_t)16 * CD;
  const bf16_t* bPanel = WoutT + (size_t)n0 * CD;

  v8f acc[8];
  #pragma unroll
  for (int i = 0; i < 8; ++i) { v8f z = {}; acc[i] = z; }

  for (int kc = 0; kc < CD; kc += 64) {
    stage_b_tile(ldsB, bPanel + kc, CD, CD - n0, tid, wave);
    __syncthreads();
    __builtin_prefetch(aBase0 + kc + 64, 0, 0);
    #pragma unroll
    for (int kl = 0; kl < 64; kl += 32) {
      v16bf a0 = load_frag_b16(aBase0 + kc + kl);
      v16bf a1 = load_frag_b16(aBase1 + kc + kl);
      #pragma unroll
      for (int nt = 0; nt < 4; ++nt) {
        v16bf bv = load_frag_b16(&ldsB[(nt * 16 + mrow) * 64 + kl + kb]);
        acc[nt * 2 + 0] = wmma_bf16(a0, bv, acc[nt * 2 + 0]);
        acc[nt * 2 + 1] = wmma_bf16(a1, bv, acc[nt * 2 + 1]);
      }
    }
    __syncthreads();
  }

  #pragma unroll
  for (int nt = 0; nt < 4; ++nt) {
    const int col = n0 + nt * 16 + (lane & 15);
    const float bias = bout[col];
    #pragma unroll
    for (int mt = 0; mt < 2; ++mt) {
      #pragma unroll
      for (int r = 0; r < 8; ++r) {
        const int row = m0 + mt * 16 + r + 8 * hf;
        out[(size_t)row * CD + col] = acc[nt * 2 + mt][r] + bias;
      }
    }
  }
}

// ---------------------------------------------------------------------------
// Host launcher.
// ---------------------------------------------------------------------------
extern "C" void kernel_launch(void* const* d_in, const int* in_sizes, int n_in,
                              void* d_out, int out_size, void* d_ws, size_t ws_size,
                              hipStream_t stream) {
  (void)in_sizes; (void)n_in; (void)out_size; (void)ws_size;
  const float* x     = (const float*)d_in[0];
  const unsigned char* smask = (const unsigned char*)d_in[1];
  const float* gamma = (const float*)d_in[2];
  const float* beta  = (const float*)d_in[3];
  const float* Wqkv  = (const float*)d_in[4];
  const float* bqkv  = (const float*)d_in[5];
  const float* Wout  = (const float*)d_in[6];
  const float* bout  = (const float*)d_in[7];
  float* out = (float*)d_out;

  const size_t MT = (size_t)CB * CT;   // 8192 rows
  char* w = (char*)d_ws;
  bf16_t* Xn    = (bf16_t*)w; w += MT * CD * 2;
  bf16_t* WqkvT = (bf16_t*)w; w += (size_t)3 * CD * CD * 2;
  bf16_t* WoutT = (bf16_t*)w; w += (size_t)CD * CD * 2;
  bf16_t* Qb    = (bf16_t*)w; w += MT * CD * 2;
  bf16_t* Kb    = (bf16_t*)w; w += MT * CD * 2;
  bf16_t* Vt    = (bf16_t*)w; w += MT * CD * 2;
  bf16_t* Obuf  = (bf16_t*)w; w += MT * CD * 2;

  {
    size_t n1 = (size_t)CD * 3 * CD;
    k_transpose_cast<<<dim3((unsigned)((n1 + 255) / 256)), dim3(256), 0, stream>>>(
        Wqkv, WqkvT, CD, 3 * CD);
    size_t n2 = (size_t)CD * CD;
    k_transpose_cast<<<dim3((unsigned)((n2 + 255) / 256)), dim3(256), 0, stream>>>(
        Wout, WoutT, CD, CD);
  }
  k_layernorm<<<dim3((unsigned)MT), dim3(256), 0, stream>>>(x, gamma, beta, Xn);
  // N=3072 -> 48 col blocks of 64; M=8192 -> 64 row blocks of 128
  k_qkv_gemm_rope<<<dim3(48, 64), dim3(128), 0, stream>>>(
      Xn, WqkvT, bqkv, Qb, Kb, Vt);
  // (T/64 q-blocks) x (B*H)
  k_attention<<<dim3(CT / 64, CB * CH), dim3(128), 0, stream>>>(
      Qb, Kb, Vt, smask, Obuf);
  // N=1024 -> 16 col blocks of 64
  k_out_gemm<<<dim3(16, 64), dim3(128), 0, stream>>>(Obuf, WoutT, bout, out);
}
